// EGCL_13692355739802
// MI455X (gfx1250) — matrix operand order repaired and checked by reading, not compiled, not run on hardware
//
#include <hip/hip_runtime.h>
#include <hip/hip_bf16.h>

typedef __attribute__((ext_vector_type(16))) __bf16 v16bf;
typedef __attribute__((ext_vector_type(8)))  float  v8f;

#define MROWS 32         // rows (edges/nodes) per wave
#define EWAVES 8         // waves per block, edge kernel (persistent)
#define NWAVES 2         // waves per block, node kernel
#define EGRID  384       // persistent edge blocks

static __device__ __forceinline__ float silu_f(float x) {
    return x / (1.0f + __expf(-x));
}

static __device__ __forceinline__ v16bf load_bfrag(const __bf16* p) {
    return *(const v16bf*)p;
}

// ---------------------------------------------------------------------------
// Pack a [K,128] f32 weight into bf16 WMMA B-fragment order.
// Fragment (kk, nt): 32 lanes x 16 bf16 contiguous.
// B layout (16-bit, 32x16): lane n = lane&15, half = lane>>4;
//   VGPR j holds K = (j<4?0:16) + half*8 + (j&3)*2, +1  (ISA 7.12.2)
// ---------------------------------------------------------------------------
__global__ void egcl_pack(const float* __restrict__ W, __bf16* __restrict__ out, int chunks) {
    int t = blockIdx.x * blockDim.x + threadIdx.x;
    int total = chunks * 4096;                       // chunks * 8 tiles * 32 lanes * 16
    if (t >= total) return;
    int elem = t & 15;
    int lane = (t >> 4) & 31;
    int nt   = (t >> 9) & 7;
    int kk   = t >> 12;
    int j = elem >> 1, s = elem & 1;
    int k = kk * 32 + ((elem >> 3) << 4) + ((lane >> 4) << 3) + (elem & 7);
    int n = (nt << 4) + (lane & 15);
    out[t] = (__bf16)W[k * 128 + n];
}

__global__ void egcl_zero(float* __restrict__ agg, float* __restrict__ fsum, int N) {
    int t = blockIdx.x * blockDim.x + threadIdx.x;
    int na = N * 128;
    if (t < na) agg[t] = 0.0f;
    else if (t < na + N * 4) fsum[t - na] = 0.0f;
}

// A-fragment K index for 16-bit A 16x32 (ISA 7.12.2):
// lane row r = lane&15, half = lane>>4; VGPR j: K = (j<4?0:16) + half*8 + (j&3)*2
static __device__ __forceinline__ int afrag_k(int j, int hi) {
    return ((j >> 2) << 4) + (hi << 3) + ((j & 3) << 1);
}

// Build an A fragment (16x32) from a per-lane row pointer (32 contiguous f32).
static __device__ __forceinline__ v16bf make_afrag_g(const float* rp, int hi) {
    v16bf a;
#pragma unroll
    for (int j = 0; j < 8; ++j) {
        int k = afrag_k(j, hi);
        float2 f = *(const float2*)(rp + k);
        a[2 * j]     = (__bf16)f.x;
        a[2 * j + 1] = (__bf16)f.y;
    }
    return a;
}

// Build an A fragment from an LDS row (128 f32), chunk kk.
static __device__ __forceinline__ v16bf make_afrag_l(const float* lrow, int kk, int hi) {
    v16bf a;
#pragma unroll
    for (int j = 0; j < 8; ++j) {
        int k = kk * 32 + afrag_k(j, hi);
        a[2 * j]     = (__bf16)lrow[k];
        a[2 * j + 1] = (__bf16)lrow[k + 1];
    }
    return a;
}

// ---------------------------------------------------------------------------
// Edge kernel: persistent, 8 waves/block, 32 edges per wave-tile.
// All edge-MLP weights (We1p|We2p|Wc1p = 128 KB packed bf16 fragments) are
// staged in LDS once per block; B reads are ds_load_b128, no loadcnt stalls
// in the WMMA chain. Dynamic LDS: 128 KB weights + 8 * 16 KB tile buffers.
// ---------------------------------------------------------------------------
__global__ __launch_bounds__(32 * EWAVES) void egcl_edge(
    const float* __restrict__ h, const float* __restrict__ coord_diff,
    const int* __restrict__ row, const int* __restrict__ col,
    const char* __restrict__ wpack,                  // We1p(64K)|We2p(32K)|Wc1p(32K)
    const float* __restrict__ We1,                   // f32, for radial row 256
    const float* __restrict__ be1, const float* __restrict__ be2,
    const float* __restrict__ bc1, const float* __restrict__ Wc2,
    float* __restrict__ agg, float* __restrict__ fsum, int E)
{
    extern __shared__ char smem[];                   // 256 KB dynamic
    const __bf16* We1l = (const __bf16*)(smem);              // 64 KB
    const __bf16* We2l = (const __bf16*)(smem + 65536);      // 32 KB
    const __bf16* Wc1l = (const __bf16*)(smem + 98304);      // 32 KB
    float* mbufAll = (float*)(smem + 131072);                // 8 * 32*128 f32

    __shared__ float sradial[EWAVES][MROWS];
    __shared__ float scoef[EWAVES][MROWS];
    __shared__ int   srow[EWAVES][MROWS];
    __shared__ int   scol[EWAVES][MROWS];
    __shared__ float scd[EWAVES][MROWS][3];

    const int tid  = threadIdx.x;
    const int wave = tid >> 5;
    const int lane = tid & 31;
    const int r    = lane & 15;
    const int hi   = lane >> 4;
    float* mbuf = mbufAll + (size_t)wave * MROWS * 128;      // [32][128]

    // ---- stage all packed edge weights into LDS (once per block) ----------
    {
        const uint4* src = (const uint4*)wpack;
        uint4* dst = (uint4*)smem;
        for (int i = tid; i < 131072 / 16; i += 32 * EWAVES) dst[i] = src[i];
    }
    __syncthreads();

    const long totalTiles = (long)(E + MROWS - 1) / MROWS;
    const int  slots   = EGRID * EWAVES;
    const int  numIter = (int)((totalTiles + slots - 1) / slots);

    const v8f zf = {};
    for (int it = 0; it < numIter; ++it) {
        const long tile = (long)it * slots + blockIdx.x * EWAVES + wave;
        const long tileBase = tile * MROWS;

        // Edge metadata: one edge per lane
        {
            long e  = tileBase + lane;
            long ec = e < E ? e : (long)(E - 1);
            srow[wave][lane] = row[ec];
            scol[wave][lane] = col[ec];
            float cx = coord_diff[ec * 3 + 0];
            float cy = coord_diff[ec * 3 + 1];
            float cz = coord_diff[ec * 3 + 2];
            scd[wave][lane][0] = cx; scd[wave][lane][1] = cy; scd[wave][lane][2] = cz;
            sradial[wave][lane] = cx * cx + cy * cy + cz * cz;
            scoef[wave][lane]   = 0.0f;
        }
        __syncthreads();

        // ---------------- GEMM1: [32 x 257] @ We1 -> hidden1 -----------------
        v8f accA[8], accB[8];
#pragma unroll
        for (int nt = 0; nt < 8; ++nt) { accA[nt] = zf; accB[nt] = zf; }
        for (int kk = 0; kk < 8; ++kk) {
            const float* rp0 = (kk < 4)
                ? (h + (size_t)srow[wave][r] * 128 + kk * 32)
                : (h + (size_t)scol[wave][r] * 128 + (kk - 4) * 32);
            const float* rp1 = (kk < 4)
                ? (h + (size_t)srow[wave][16 + r] * 128 + kk * 32)
                : (h + (size_t)scol[wave][16 + r] * 128 + (kk - 4) * 32);
            v16bf a0 = make_afrag_g(rp0, hi);
            v16bf a1 = make_afrag_g(rp1, hi);
#pragma unroll
            for (int nt = 0; nt < 8; ++nt) {
                v16bf b = load_bfrag(We1l + (((size_t)kk * 8 + nt) * 32 + lane) * 16);
                accA[nt] = __builtin_amdgcn_wmma_f32_16x16x32_bf16(false, a0, false, b,
                                                                   (short)0, accA[nt], false, false);
                accB[nt] = __builtin_amdgcn_wmma_f32_16x16x32_bf16(false, a1, false, b,
                                                                   (short)0, accB[nt], false, false);
            }
        }
#pragma unroll
        for (int nt = 0; nt < 8; ++nt) {
            int n = nt * 16 + r;
            float w256 = We1[256 * 128 + n];
            float b1   = be1[n];
#pragma unroll
            for (int j = 0; j < 8; ++j) {
                int M = j + hi * 8;
                mbuf[M * 128 + n]        = silu_f(accA[nt][j] + sradial[wave][M] * w256 + b1);
                mbuf[(M + 16) * 128 + n] = silu_f(accB[nt][j] + sradial[wave][M + 16] * w256 + b1);
            }
        }
        __syncthreads();

        // ---------------- GEMM2: hidden1 @ We2 -> m ---------------------------
#pragma unroll
        for (int nt = 0; nt < 8; ++nt) { accA[nt] = zf; accB[nt] = zf; }
        for (int kk = 0; kk < 4; ++kk) {
            v16bf a0 = make_afrag_l(mbuf + r * 128,        kk, hi);
            v16bf a1 = make_afrag_l(mbuf + (16 + r) * 128, kk, hi);
#pragma unroll
            for (int nt = 0; nt < 8; ++nt) {
                v16bf b = load_bfrag(We2l + (((size_t)kk * 8 + nt) * 32 + lane) * 16);
                accA[nt] = __builtin_amdgcn_wmma_f32_16x16x32_bf16(false, a0, false, b,
                                                                   (short)0, accA[nt], false, false);
                accB[nt] = __builtin_amdgcn_wmma_f32_16x16x32_bf16(false, a1, false, b,
                                                                   (short)0, accB[nt], false, false);
            }
        }
        __syncthreads();   // hidden1 reads done before overwrite
#pragma unroll
        for (int nt = 0; nt < 8; ++nt) {
            int n = nt * 16 + r;
            float b2 = be2[n];
#pragma unroll
            for (int j = 0; j < 8; ++j) {
                int M = j + hi * 8;
                mbuf[M * 128 + n]        = silu_f(accA[nt][j] + b2);   // m
                mbuf[(M + 16) * 128 + n] = silu_f(accB[nt][j] + b2);
            }
        }
        __syncthreads();

        // scatter agg += m (global f32 atomics; target is L2-resident)
        for (int t = lane; t < MROWS * 128; t += 32) {
            int le = t >> 7, c = t & 127;
            long e = tileBase + le;
            if (e < E)
                unsafeAtomicAdd(&agg[(size_t)srow[wave][le] * 128 + c], mbuf[le * 128 + c]);
        }

        // ---------------- GEMM3: m @ Wc1 -> silu -> dot Wc2 -------------------
#pragma unroll
        for (int nt = 0; nt < 8; ++nt) { accA[nt] = zf; accB[nt] = zf; }
        for (int kk = 0; kk < 4; ++kk) {
            v16bf a0 = make_afrag_l(mbuf + r * 128,        kk, hi);
            v16bf a1 = make_afrag_l(mbuf + (16 + r) * 128, kk, hi);
#pragma unroll
            for (int nt = 0; nt < 8; ++nt) {
                v16bf b = load_bfrag(Wc1l + (((size_t)kk * 8 + nt) * 32 + lane) * 16);
                accA[nt] = __builtin_amdgcn_wmma_f32_16x16x32_bf16(false, a0, false, b,
                                                                   (short)0, accA[nt], false, false);
                accB[nt] = __builtin_amdgcn_wmma_f32_16x16x32_bf16(false, a1, false, b,
                                                                   (short)0, accB[nt], false, false);
            }
        }
#pragma unroll
        for (int nt = 0; nt < 8; ++nt) {
            int n = nt * 16 + r;
            float bc = bc1[n];
            float wc = Wc2[n];
#pragma unroll
            for (int j = 0; j < 8; ++j) {
                int M = j + hi * 8;
                atomicAdd(&scoef[wave][M],      silu_f(accA[nt][j] + bc) * wc);  // ds_add_f32
                atomicAdd(&scoef[wave][M + 16], silu_f(accB[nt][j] + bc) * wc);
            }
        }
        __syncthreads();

        // force accumulation (clamped trans, scatter-sum + count); one edge/lane
        {
            long e = tileBase + lane;
            if (e < E) {
                float cf = scoef[wave][lane];
                int ri = srow[wave][lane];
#pragma unroll
                for (int k = 0; k < 3; ++k) {
                    float t = scd[wave][lane][k] * cf;
                    t = fminf(fmaxf(t, -100.0f), 100.0f);
                    unsafeAtomicAdd(&fsum[(size_t)ri * 4 + k], t);
                }
                unsafeAtomicAdd(&fsum[(size_t)ri * 4 + 3], 1.0f);
            }
        }
        __syncthreads();
    }
}

// ---------------------------------------------------------------------------
// Node kernel: 2 waves/block, 32 nodes per wave.
// ---------------------------------------------------------------------------
__global__ __launch_bounds__(32 * NWAVES) void egcl_node(
    const float* __restrict__ h, const float* __restrict__ agg,
    const float* __restrict__ fsum,
    const __bf16* __restrict__ Wn1p, const float* __restrict__ bn1,
    const __bf16* __restrict__ Wn2p, const float* __restrict__ bn2,
    const __bf16* __restrict__ Wv1p, const float* __restrict__ bv1,
    const float* __restrict__ Wv2, const float* __restrict__ bv2,
    float* __restrict__ out, int N)
{
    __shared__ float mbuf[NWAVES][MROWS][128];
    __shared__ float scoef[NWAVES][MROWS];

    const int tid  = threadIdx.x;
    const int wave = tid >> 5;
    const int lane = tid & 31;
    const int r    = lane & 15;
    const int hi   = lane >> 4;

    int tiles = (N + MROWS - 1) / MROWS;
    int tile  = blockIdx.x * NWAVES + wave;
    if (tile >= tiles) tile = tiles - 1;       // duplicate-recompute, uniform per wave
    const long base = (long)tile * MROWS;

    scoef[wave][lane] = 0.0f;
    __syncthreads();

    const v8f zf = {};
    // ---------------- node GEMM1: [h | agg] @ Wn1 ----------------------------
    v8f accA[8], accB[8];
#pragma unroll
    for (int nt = 0; nt < 8; ++nt) { accA[nt] = zf; accB[nt] = zf; }
    for (int kk = 0; kk < 8; ++kk) {
        const float* rp0 = (kk < 4)
            ? (h   + (size_t)(base + r) * 128 + kk * 32)
            : (agg + (size_t)(base + r) * 128 + (kk - 4) * 32);
        const float* rp1 = (kk < 4)
            ? (h   + (size_t)(base + 16 + r) * 128 + kk * 32)
            : (agg + (size_t)(base + 16 + r) * 128 + (kk - 4) * 32);
        v16bf a0 = make_afrag_g(rp0, hi);
        v16bf a1 = make_afrag_g(rp1, hi);
#pragma unroll
        for (int nt = 0; nt < 8; ++nt) {
            v16bf b = load_bfrag(Wn1p + (((size_t)kk * 8 + nt) * 32 + lane) * 16);
            accA[nt] = __builtin_amdgcn_wmma_f32_16x16x32_bf16(false, a0, false, b,
                                                               (short)0, accA[nt], false, false);
            accB[nt] = __builtin_amdgcn_wmma_f32_16x16x32_bf16(false, a1, false, b,
                                                               (short)0, accB[nt], false, false);
        }
    }
#pragma unroll
    for (int nt = 0; nt < 8; ++nt) {
        int n = nt * 16 + r;
        float b1 = bn1[n];
#pragma unroll
        for (int j = 0; j < 8; ++j) {
            int M = j + hi * 8;
            mbuf[wave][M][n]      = silu_f(accA[nt][j] + b1);
            mbuf[wave][M + 16][n] = silu_f(accB[nt][j] + b1);
        }
    }
    __syncthreads();

    // ---------------- node GEMM2: u @ Wn2 + bn2 -> h_out ---------------------
#pragma unroll
    for (int nt = 0; nt < 8; ++nt) { accA[nt] = zf; accB[nt] = zf; }
    for (int kk = 0; kk < 4; ++kk) {
        v16bf a0 = make_afrag_l(&mbuf[wave][r][0],      kk, hi);
        v16bf a1 = make_afrag_l(&mbuf[wave][16 + r][0], kk, hi);
#pragma unroll
        for (int nt = 0; nt < 8; ++nt) {
            v16bf b = load_bfrag(Wn2p + (((size_t)kk * 8 + nt) * 32 + lane) * 16);
            accA[nt] = __builtin_amdgcn_wmma_f32_16x16x32_bf16(false, a0, false, b,
                                                               (short)0, accA[nt], false, false);
            accB[nt] = __builtin_amdgcn_wmma_f32_16x16x32_bf16(false, a1, false, b,
                                                               (short)0, accB[nt], false, false);
        }
    }
    // out layout: [vel N][force 3N][h_out 128N]
    float* hout = out + (size_t)4 * N;
#pragma unroll
    for (int nt = 0; nt < 8; ++nt) {
        int n = nt * 16 + r;
        float b2 = bn2[n];
#pragma unroll
        for (int j = 0; j < 8; ++j) {
            long n0 = base + j + hi * 8;
            long n1 = n0 + 16;
            if (n0 < N) hout[(size_t)n0 * 128 + n] = accA[nt][j] + b2;
            if (n1 < N) hout[(size_t)n1 * 128 + n] = accB[nt][j] + b2;
        }
    }

    // ---------------- velocity head: silu(h @ Wv1 + bv1) . Wv2 + bv2 ---------
#pragma unroll
    for (int nt = 0; nt < 8; ++nt) { accA[nt] = zf; accB[nt] = zf; }
    for (int kk = 0; kk < 4; ++kk) {
        v16bf a0 = make_afrag_g(h + (size_t)(base + r) * 128 + kk * 32, hi);
        v16bf a1 = make_afrag_g(h + (size_t)(base + 16 + r) * 128 + kk * 32, hi);
#pragma unroll
        for (int nt = 0; nt < 8; ++nt) {
            v16bf b = load_bfrag(Wv1p + (((size_t)kk * 8 + nt) * 32 + lane) * 16);
            accA[nt] = __builtin_amdgcn_wmma_f32_16x16x32_bf16(false, a0, false, b,
                                                               (short)0, accA[nt], false, false);
            accB[nt] = __builtin_amdgcn_wmma_f32_16x16x32_bf16(false, a1, false, b,
                                                               (short)0, accB[nt], false, false);
        }
    }
#pragma unroll
    for (int nt = 0; nt < 8; ++nt) {
        int n = nt * 16 + r;
        float bv = bv1[n];
        float wv = Wv2[n];
#pragma unroll
        for (int j = 0; j < 8; ++j) {
            int M = j + hi * 8;
            atomicAdd(&scoef[wave][M],      silu_f(accA[nt][j] + bv) * wv);
            atomicAdd(&scoef[wave][M + 16], silu_f(accB[nt][j] + bv) * wv);
        }
    }
    __syncthreads();

    {
        long node = base + lane;
        if (node < N) {
            out[node] = scoef[wave][lane] + bv2[0];            // vel
            float cnt = fmaxf(fsum[(size_t)node * 4 + 3], 1.0f);
#pragma unroll
            for (int k = 0; k < 3; ++k)
                out[(size_t)N + node * 3 + k] = fsum[(size_t)node * 4 + k] / cnt;
        }
    }
}

// ---------------------------------------------------------------------------
extern "C" void kernel_launch(void* const* d_in, const int* in_sizes, int n_in,
                              void* d_out, int out_size, void* d_ws, size_t ws_size,
                              hipStream_t stream) {
    const float* h   = (const float*)d_in[0];
    const float* cd  = (const float*)d_in[1];
    const int*   row = (const int*)d_in[2];
    const int*   col = (const int*)d_in[3];
    const float* We1 = (const float*)d_in[4];
    const float* be1 = (const float*)d_in[5];
    const float* We2 = (const float*)d_in[6];
    const float* be2 = (const float*)d_in[7];
    const float* Wn1 = (const float*)d_in[8];
    const float* bn1 = (const float*)d_in[9];
    const float* Wn2 = (const float*)d_in[10];
    const float* bn2 = (const float*)d_in[11];
    const float* Wc1 = (const float*)d_in[12];
    const float* bc1 = (const float*)d_in[13];
    const float* Wc2 = (const float*)d_in[14];
    const float* Wv1 = (const float*)d_in[15];
    const float* bv1 = (const float*)d_in[16];
    const float* Wv2 = (const float*)d_in[17];
    const float* bv2 = (const float*)d_in[18];

    const int N = in_sizes[0] / 128;
    const int E = in_sizes[2];

    char* ws = (char*)d_ws;
    __bf16* We1p = (__bf16*)(ws + 0);         // 8 chunks -> 64 KB   (contiguous
    __bf16* We2p = (__bf16*)(ws + 65536);     // 4 chunks -> 32 KB    128 KB edge
    __bf16* Wc1p = (__bf16*)(ws + 98304);     // 32 KB                weight pack)
    __bf16* Wn1p = (__bf16*)(ws + 131072);    // 64 KB
    __bf16* Wn2p = (__bf16*)(ws + 196608);    // 32 KB
    __bf16* Wv1p = (__bf16*)(ws + 229376);    // 32 KB
    float*  agg  = (float*)(ws + 262144);                       // N*128 f32
    float*  fsum = (float*)(ws + 262144 + (size_t)N * 128 * 4); // N*4 f32

    // pack weights into WMMA B-fragment order (deterministic, cheap)
    egcl_pack<<<(8 * 4096 + 255) / 256, 256, 0, stream>>>(We1, We1p, 8);
    egcl_pack<<<(4 * 4096 + 255) / 256, 256, 0, stream>>>(We2, We2p, 4);
    egcl_pack<<<(4 * 4096 + 255) / 256, 256, 0, stream>>>(Wc1, Wc1p, 4);
    egcl_pack<<<(8 * 4096 + 255) / 256, 256, 0, stream>>>(Wn1, Wn1p, 8);
    egcl_pack<<<(4 * 4096 + 255) / 256, 256, 0, stream>>>(Wn2, Wn2p, 4);
    egcl_pack<<<(4 * 4096 + 255) / 256, 256, 0, stream>>>(Wv1, Wv1p, 4);

    int zt = N * 132;
    egcl_zero<<<(zt + 255) / 256, 256, 0, stream>>>(agg, fsum, N);

    // persistent edge kernel: 256 KB dynamic LDS (128 KB weights + 8 tile bufs)
    const int edyn = 131072 + EWAVES * MROWS * 128 * 4;   // 262144 B
    hipFuncSetAttribute((const void*)egcl_edge,
                        hipFuncAttributeMaxDynamicSharedMemorySize, edyn);
    egcl_edge<<<EGRID, 32 * EWAVES, edyn, stream>>>(
        h, cd, row, col, (const char*)ws, We1, be1, be2, bc1, Wc2, agg, fsum, E);

    int tiles = (N + MROWS - 1) / MROWS;
    int nblocks = (tiles + NWAVES - 1) / NWAVES;
    egcl_node<<<nblocks, 32 * NWAVES, 0, stream>>>(
        h, agg, fsum, Wn1p, bn1, Wn2p, bn2, Wv1p, bv1, Wv2, bv2, (float*)d_out, N);
}
